// GatedMoE_30949534335418
// MI455X (gfx1250) — matrix operand order repaired and checked by reading, not compiled, not run on hardware
//
#include <hip/hip_runtime.h>
#include <stdint.h>

// ---------------------------------------------------------------------------
// Gated MoE (top-2 of 8 experts), routed grouped-GEMM for MI455X (gfx1250).
// bf16 WMMA 16x16x32, f32 accumulate, LDS double-buffered tiles fed by the
// CDNA5 async global->LDS engine (ASYNCcnt) when available.
// ---------------------------------------------------------------------------

typedef __attribute__((ext_vector_type(16))) __bf16 v16bf;
typedef __attribute__((ext_vector_type(8)))  __bf16 v8bf;
typedef __attribute__((ext_vector_type(8)))  float  v8f;

typedef int vi4 __attribute__((vector_size(16)));
typedef __attribute__((address_space(1))) vi4* gvi4p;   // global ptr to 16B vector
typedef __attribute__((address_space(3))) vi4* lvi4p;   // LDS ptr to 16B vector

constexpr int TOK = 8192;   // B*N tokens
constexpr int DIM = 1024;   // D
constexpr int FF  = 4096;   // F
constexpr int NE  = 8;      // experts
constexpr int ASSIGN = TOK * 2;

constexpr int KC   = 32;    // k elements staged per chunk (one WMMA K)
constexpr int LDP  = 40;    // padded LDS row (elements): 20 dwords -> conflict-free

// workspace layout (bytes)
constexpr size_t O_SCORE = 0;                                              // [TOK*2] f32 gate scores
constexpr size_t O_CNT   = O_SCORE + size_t(TOK) * 2 * 4;                  // [NE] i32
constexpr size_t O_LIST  = (O_CNT + NE * 4 + 255) & ~size_t(255);          // [NE][TOK] i32 (entry = t<<1 | k)
constexpr size_t O_XB    = (O_LIST + size_t(NE) * TOK * 4 + 255) & ~size_t(255);   // x in bf16
constexpr size_t O_W1T   = (O_XB  + size_t(TOK) * DIM * 2 + 255) & ~size_t(255);   // W1^T [e][f][d] bf16
constexpr size_t O_W2T   = (O_W1T + size_t(NE) * FF * DIM * 2 + 255) & ~size_t(255); // W2^T [e][d][f] bf16
constexpr size_t O_H     = (O_W2T + size_t(NE) * DIM * FF * 2 + 255) & ~size_t(255); // h [ASSIGN][FF] bf16
constexpr size_t WS_NEED = O_H + size_t(ASSIGN) * FF * 2;                  // ~272 MB

// ------------------------- async global->LDS helpers -----------------------

#if defined(__has_builtin)
#if __has_builtin(__builtin_amdgcn_global_load_async_to_lds_b128)
#define HAVE_ASYNC_LDS 1
#endif
#endif

__device__ __forceinline__ void cp_b128(const uint16_t* g, uint16_t* l) {
#ifdef HAVE_ASYNC_LDS
    __builtin_amdgcn_global_load_async_to_lds_b128(
        (gvi4p)(uint16_t*)g, (lvi4p)l, 0, 0);
#else
    *(v8bf*)l = *(const v8bf*)g;   // sync fallback: global_load + ds_store
#endif
}

template <int N>
__device__ __forceinline__ void async_wait() {
#ifdef HAVE_ASYNC_LDS
    asm volatile("s_wait_asynccnt %0" :: "i"(N) : "memory");
#endif
}

__device__ __forceinline__ uint16_t f2bf_bits(float f) {
    union { float f; uint32_t u; } v; v.f = f;
    uint32_t u = v.u;
    if ((u & 0x7fffffffu) > 0x7f800000u) return (uint16_t)((u >> 16) | 0x40); // quiet NaN
    uint32_t r = u + 0x7fffu + ((u >> 16) & 1u);                              // RNE
    return (uint16_t)(r >> 16);
}

// ------------------------------- utility kernels ---------------------------

__global__ void zero_f32(float* p, int n) {
    int i = blockIdx.x * 256 + threadIdx.x;
    if (i < n) p[i] = 0.0f;
}
__global__ void zero_i32(int* p, int n) {
    int i = blockIdx.x * 32 + threadIdx.x;
    if (i < n) p[i] = 0;
}
__global__ void convert_x(const float* __restrict__ x, uint16_t* __restrict__ xb, int n) {
    int i = blockIdx.x * 256 + threadIdx.x;
    if (i < n) xb[i] = f2bf_bits(x[i]);
}

// [Z][R][C] f32 -> [Z][C][R] bf16, 32x32 tiles via LDS. block (32,8).
__global__ void transpose_cvt(const float* __restrict__ src, uint16_t* __restrict__ dst,
                              int R, int C) {
    __shared__ uint16_t tile[32][33];
    const int z  = blockIdx.z;
    const int c0 = blockIdx.x * 32, r0 = blockIdx.y * 32;
    const float*  s = src + size_t(z) * R * C;
    uint16_t*     d = dst + size_t(z) * R * C;
    const int tx = threadIdx.x, ty = threadIdx.y;
#pragma unroll
    for (int j = 0; j < 32; j += 8)
        tile[ty + j][tx] = f2bf_bits(s[size_t(r0 + ty + j) * C + c0 + tx]);
    __syncthreads();
#pragma unroll
    for (int j = 0; j < 32; j += 8)
        d[size_t(c0 + ty + j) * R + r0 + tx] = tile[tx][ty + j];
}

// ------------------------------- gating ------------------------------------

__global__ __launch_bounds__(256) void gating(const float* __restrict__ x,
                                              const float* __restrict__ Wg,
                                              const float* __restrict__ bg,
                                              float* __restrict__ score,
                                              int* __restrict__ counts,
                                              int* __restrict__ list) {
    const int wave = threadIdx.x >> 5, lane = threadIdx.x & 31;
    const int t = blockIdx.x * 8 + wave;
    float acc[NE] = {};
    const float* xr = x + size_t(t) * DIM;
    for (int d = lane; d < DIM; d += 32) {
        float xv = xr[d];
#pragma unroll
        for (int e = 0; e < NE; e++) acc[e] += xv * Wg[d * NE + e];
    }
#pragma unroll
    for (int e = 0; e < NE; e++) {
#pragma unroll
        for (int off = 16; off >= 1; off >>= 1)
            acc[e] += __shfl_xor(acc[e], off, 32);
    }
    if (lane == 0) {
        float l[NE];
#pragma unroll
        for (int e = 0; e < NE; e++) l[e] = acc[e] + bg[e];
        int i1 = 0;
#pragma unroll
        for (int e = 1; e < NE; e++) if (l[e] > l[i1]) i1 = e;
        int i2 = (i1 == 0) ? 1 : 0;
#pragma unroll
        for (int e = 0; e < NE; e++) if (e != i1 && l[e] > l[i2]) i2 = e;
        float mx = l[i1], sum = 0.f;
#pragma unroll
        for (int e = 0; e < NE; e++) sum += __expf(l[e] - mx);
        score[t * 2 + 0] = 1.0f / sum;
        score[t * 2 + 1] = __expf(l[i2] - mx) / sum;
        int s0 = atomicAdd(&counts[i1], 1); list[i1 * TOK + s0] = (t << 1);
        int s1 = atomicAdd(&counts[i2], 1); list[i2 * TOK + s1] = (t << 1) | 1;
    }
}

// ------------------------------- grouped GEMM 1 ----------------------------
// h[entry] = relu(Xg @ W1[e] + b1[e]).  128x128 tile, 8 waves (4M x 2N),
// 2x4 WMMA C frags per wave.  Tiles staged in double-buffered LDS via async
// global->LDS; fragments read with ds_load_b128 (padded rows, conflict-free).
__global__ __launch_bounds__(256) void moe_gemm1(const uint16_t* __restrict__ Xb,
                                                 const uint16_t* __restrict__ W1t,
                                                 const float* __restrict__ b1,
                                                 const int* __restrict__ counts,
                                                 const int* __restrict__ list,
                                                 uint16_t* __restrict__ H) {
    __shared__ uint16_t As[2][128][LDP];
    __shared__ uint16_t Bs[2][128][LDP];

    const int e   = blockIdx.z;
    const int cnt = counts[e];
    const int M0  = blockIdx.y * 128;
    if (M0 >= cnt) return;
    const int N0  = blockIdx.x * 128;
    const int tid = threadIdx.x;
    const int wid = tid >> 5, lane = tid & 31;
    const int wm = wid & 3, wn = wid >> 2;
    const int ln15 = lane & 15, lhi = lane >> 4;
    const int ka = lhi * 8, kb = lhi * 16;

    // staging: thread covers row (tid>>1), 16-element half (tid&1)
    const int ra  = tid >> 1;
    const int ea0 = (tid & 1) * 16;
    int aslot = M0 + ra; aslot = aslot < cnt ? aslot : cnt - 1;
    const uint16_t* agp = Xb  + size_t(list[e * TOK + aslot] >> 1) * DIM + ea0;
    const uint16_t* bgp = W1t + size_t(e * FF + N0 + ra) * DIM + ea0;

    auto stage = [&](int kc, int buf) {
#pragma unroll
        for (int q = 0; q < 2; q++) cp_b128(agp + kc + q * 8, &As[buf][ra][ea0 + q * 8]);
#pragma unroll
        for (int q = 0; q < 2; q++) cp_b128(bgp + kc + q * 8, &Bs[buf][ra][ea0 + q * 8]);
    };

    v8f acc[2][4] = {};
    stage(0, 0);
    constexpr int NCH = DIM / KC;
    for (int c = 0; c < NCH; ++c) {
        if (c + 1 < NCH) { stage((c + 1) * KC, (c + 1) & 1); async_wait<4>(); }
        else             { async_wait<0>(); }
        __syncthreads();
        const int buf = c & 1;
        v16bf a[2];
#pragma unroll
        for (int i = 0; i < 2; i++) {
            union { v16bf v; v8bf h[2]; } u;
            u.h[0] = *(const v8bf*)&As[buf][wm * 32 + i * 16 + ln15][ka];
            u.h[1] = *(const v8bf*)&As[buf][wm * 32 + i * 16 + ln15][16 + ka];
            a[i] = u.v;
        }
#pragma unroll
        for (int j = 0; j < 4; j++) {
            union { v16bf v; v8bf h[2]; } u;
            u.h[0] = *(const v8bf*)&Bs[buf][wn * 64 + j * 16 + ln15][ka];
            u.h[1] = *(const v8bf*)&Bs[buf][wn * 64 + j * 16 + ln15][16 + ka];
#pragma unroll
            for (int i = 0; i < 2; i++)
                acc[i][j] = __builtin_amdgcn_wmma_f32_16x16x32_bf16(
                    false, a[i], false, u.v, (short)0, acc[i][j], false, false);
        }
        __syncthreads();
    }

    const int mwb = M0 + wm * 32, nwb = N0 + wn * 64;
    int ent[2][8];
#pragma unroll
    for (int i = 0; i < 2; i++)
#pragma unroll
        for (int r = 0; r < 8; r++) {
            int slot = mwb + i * 16 + lhi * 8 + r;
            ent[i][r] = (slot < cnt) ? list[e * TOK + slot] : -1;
        }
#pragma unroll
    for (int j = 0; j < 4; j++) {
        const int f = nwb + j * 16 + ln15;
        const float bias = b1[e * FF + f];
#pragma unroll
        for (int i = 0; i < 2; i++)
#pragma unroll
            for (int r = 0; r < 8; r++)
                if (ent[i][r] >= 0) {
                    float v = acc[i][j][r] + bias;
                    v = v > 0.f ? v : 0.f;
                    H[size_t(ent[i][r]) * FF + f] = f2bf_bits(v);
                }
    }
}

// ------------------------------- grouped GEMM 2 ----------------------------
// out[t] += score(entry) * (h[entry] @ W2[e] + b2[e]); exactly 2 commutative
// f32 atomic adds per output element -> deterministic.
__global__ __launch_bounds__(256) void moe_gemm2(const uint16_t* __restrict__ H,
                                                 const uint16_t* __restrict__ W2t,
                                                 const float* __restrict__ b2,
                                                 const float* __restrict__ score,
                                                 const int* __restrict__ counts,
                                                 const int* __restrict__ list,
                                                 float* __restrict__ out) {
    __shared__ uint16_t As[2][128][LDP];
    __shared__ uint16_t Bs[2][128][LDP];

    const int e   = blockIdx.z;
    const int cnt = counts[e];
    const int M0  = blockIdx.y * 128;
    if (M0 >= cnt) return;
    const int N0  = blockIdx.x * 128;
    const int tid = threadIdx.x;
    const int wid = tid >> 5, lane = tid & 31;
    const int wm = wid & 3, wn = wid >> 2;
    const int ln15 = lane & 15, lhi = lane >> 4;
    const int ka = lhi * 8, kb = lhi * 16;

    const int ra  = tid >> 1;
    const int ea0 = (tid & 1) * 16;
    int aslot = M0 + ra; aslot = aslot < cnt ? aslot : cnt - 1;
    const uint16_t* agp = H   + size_t(list[e * TOK + aslot]) * FF + ea0;
    const uint16_t* bgp = W2t + size_t(e * DIM + N0 + ra) * FF + ea0;

    auto stage = [&](int kc, int buf) {
#pragma unroll
        for (int q = 0; q < 2; q++) cp_b128(agp + kc + q * 8, &As[buf][ra][ea0 + q * 8]);
#pragma unroll
        for (int q = 0; q < 2; q++) cp_b128(bgp + kc + q * 8, &Bs[buf][ra][ea0 + q * 8]);
    };

    v8f acc[2][4] = {};
    stage(0, 0);
    constexpr int NCH = FF / KC;
    for (int c = 0; c < NCH; ++c) {
        if (c + 1 < NCH) { stage((c + 1) * KC, (c + 1) & 1); async_wait<4>(); }
        else             { async_wait<0>(); }
        __syncthreads();
        const int buf = c & 1;
        v16bf a[2];
#pragma unroll
        for (int i = 0; i < 2; i++) {
            union { v16bf v; v8bf h[2]; } u;
            u.h[0] = *(const v8bf*)&As[buf][wm * 32 + i * 16 + ln15][ka];
            u.h[1] = *(const v8bf*)&As[buf][wm * 32 + i * 16 + ln15][16 + ka];
            a[i] = u.v;
        }
#pragma unroll
        for (int j = 0; j < 4; j++) {
            union { v16bf v; v8bf h[2]; } u;
            u.h[0] = *(const v8bf*)&Bs[buf][wn * 64 + j * 16 + ln15][ka];
            u.h[1] = *(const v8bf*)&Bs[buf][wn * 64 + j * 16 + ln15][16 + ka];
#pragma unroll
            for (int i = 0; i < 2; i++)
                acc[i][j] = __builtin_amdgcn_wmma_f32_16x16x32_bf16(
                    false, a[i], false, u.v, (short)0, acc[i][j], false, false);
        }
        __syncthreads();
    }

    const int mwb = M0 + wm * 32, nwb = N0 + wn * 64;
    int ent[2][8];
#pragma unroll
    for (int i = 0; i < 2; i++)
#pragma unroll
        for (int r = 0; r < 8; r++) {
            int slot = mwb + i * 16 + lhi * 8 + r;
            ent[i][r] = (slot < cnt) ? list[e * TOK + slot] : -1;
        }
#pragma unroll
    for (int j = 0; j < 4; j++) {
        const int dcol = nwb + j * 16 + ln15;
        const float bias = b2[e * DIM + dcol];
#pragma unroll
        for (int i = 0; i < 2; i++)
#pragma unroll
            for (int r = 0; r < 8; r++)
                if (ent[i][r] >= 0) {
                    const int entry = ent[i][r];
                    float v = (acc[i][j][r] + bias) * score[entry];
                    atomicAdd(&out[size_t(entry >> 1) * DIM + dcol], v);
                }
    }
}

// ------------------------------- launch ------------------------------------

extern "C" void kernel_launch(void* const* d_in, const int* in_sizes, int n_in,
                              void* d_out, int out_size, void* d_ws, size_t ws_size,
                              hipStream_t stream) {
    const float* x  = (const float*)d_in[0];
    const float* Wg = (const float*)d_in[1];
    const float* bg = (const float*)d_in[2];
    const float* W1 = (const float*)d_in[3];
    const float* b1 = (const float*)d_in[4];
    const float* W2 = (const float*)d_in[5];
    const float* b2 = (const float*)d_in[6];
    float* out = (float*)d_out;

    if (ws_size < WS_NEED) return;  // insufficient scratch (need ~272 MB)

    char* ws = (char*)d_ws;
    float*    score  = (float*)(ws + O_SCORE);
    int*      counts = (int*)(ws + O_CNT);
    int*      list   = (int*)(ws + O_LIST);
    uint16_t* Xb     = (uint16_t*)(ws + O_XB);
    uint16_t* W1t    = (uint16_t*)(ws + O_W1T);
    uint16_t* W2t    = (uint16_t*)(ws + O_W2T);
    uint16_t* H      = (uint16_t*)(ws + O_H);

    zero_i32<<<1, 32, 0, stream>>>(counts, NE);
    convert_x<<<(TOK * DIM) / 256, 256, 0, stream>>>(x, Xb, TOK * DIM);
    transpose_cvt<<<dim3(FF / 32, DIM / 32, NE), dim3(32, 8), 0, stream>>>(W1, W1t, DIM, FF);
    transpose_cvt<<<dim3(DIM / 32, FF / 32, NE), dim3(32, 8), 0, stream>>>(W2, W2t, FF, DIM);
    gating<<<TOK / 8, 256, 0, stream>>>(x, Wg, bg, score, counts, list);
    zero_f32<<<(TOK * DIM) / 256, 256, 0, stream>>>(out, TOK * DIM);
    moe_gemm1<<<dim3(FF / 128, TOK / 128, NE), 256, 0, stream>>>(Xb, W1t, b1, counts, list, H);
    moe_gemm2<<<dim3(DIM / 128, TOK / 128, NE), 256, 0, stream>>>(H, W2t, b2, score, counts, list, out);
}
